// BaichuanAttention_10685878632598
// MI455X (gfx1250) — matrix operand order repaired and checked by reading, not compiled, not run on hardware
//
#include <hip/hip_runtime.h>
#include <hip/hip_bf16.h>

// ---------------------------------------------------------------------------
// Baichuan attention forward for MI455X (gfx1250, wave32, WMMA).
// Pipeline: cast f32->bf16  ->  QKV GEMM (wmma bf16)  ->  flash-style causal
// attention computed TRANSPOSED (S^T = K Q^T, O^T = V^T P^T) so softmax rows
// are per-lane  ->  o_proj GEMM (wmma bf16, f32 out).
// ---------------------------------------------------------------------------

typedef __attribute__((ext_vector_type(16))) __bf16       v16bf;
typedef __attribute__((ext_vector_type(8)))  float        v8f;
typedef __attribute__((ext_vector_type(4)))  unsigned int u32x4;
typedef __attribute__((ext_vector_type(2)))  unsigned int u32x2;
typedef __attribute__((ext_vector_type(4)))  float        f32x4;
typedef __attribute__((ext_vector_type(4)))  __bf16       bf16x4;

#define HID   4096
#define NHEAD 32
#define HD    128
#define SEQ   2048
#define L2E   1.4426950408889634f
#define NEGF  (-3.0e38f)
#define QSCALE 0.088388347648318447f   /* 1/sqrt(128) */

union frag_cast { u32x4 q[2]; v16bf v; };

__device__ __forceinline__ v8f vzero8() {
  v8f z;
  #pragma unroll
  for (int i = 0; i < 8; ++i) z[i] = 0.0f;
  return z;
}

__device__ __forceinline__ v8f wmma_bf16(v16bf a, v16bf b, v8f c) {
  // D = A(16x32) * B(32x16) + C, f32 accumulate
  return __builtin_amdgcn_wmma_f32_16x16x32_bf16(false, a, false, b,
                                                 (short)0, c, false, false);
}

// A-fragment 16x32 (rows natural). Lane l: row l&15; K = {16h .. 16h+15} contiguous
__device__ __forceinline__ v16bf load_fragA(const __bf16* p, int stride) {
  const int l = threadIdx.x & 31;
  const __bf16* q = p + (size_t)(l & 15) * stride + ((l >> 4) << 4);
  frag_cast u;
  u.q[0] = *(const u32x4*)(q);
  u.q[1] = *(const u32x4*)(q + 8);
  return u.v;
}

// A-fragment with row slots mapped to rows {0..7,16..23} (slot r -> r + (r&8)).
// Lets the S^T D-tile feed the P^T B-fragment with no cross-lane movement.
__device__ __forceinline__ v16bf load_fragA_perm(const __bf16* p, int stride) {
  const int l = threadIdx.x & 31;
  const int r = l & 15;
  const __bf16* q = p + (size_t)(r + (r & 8)) * stride + ((l >> 4) << 4);
  frag_cast u;
  u.q[0] = *(const u32x4*)(q);
  u.q[1] = *(const u32x4*)(q + 8);
  return u.v;
}

// B-fragment 32x16 from rows of the transposed operand (columns of B).
// Lane l: column l&15; K = {16h .. 16h+15} contiguous -> one 32B load.
__device__ __forceinline__ v16bf load_fragB(const __bf16* p, int stride) {
  const int l = threadIdx.x & 31;
  const __bf16* q = p + (size_t)(l & 15) * stride + ((l >> 4) << 4);
  frag_cast u;
  u.q[0] = *(const u32x4*)(q);
  u.q[1] = *(const u32x4*)(q + 8);
  return u.v;
}

// ---------------------------------------------------------------------------
__global__ void __launch_bounds__(256)
cast_f32_bf16(const float* __restrict__ s, __bf16* __restrict__ d, int n4) {
  int i = blockIdx.x * blockDim.x + threadIdx.x;
  if (i >= n4) return;
  f32x4 x = *(const f32x4*)(s + (size_t)i * 4);
  bf16x4 y;
  #pragma unroll
  for (int j = 0; j < 4; ++j) y[j] = (__bf16)x[j];
  *(bf16x4*)(d + (size_t)i * 4) = y;
}

// ---------------------------------------------------------------------------
// QKV GEMM: [4096 tok x 4096] * W_pack^T -> Q[b][h][s][d] (scaled), K[b][h][s][d],
// V transposed [b][h][d][s].  Wave tile 32M x 64N, K-step 32.
__global__ void __launch_bounds__(256)
qkv_gemm_kernel(const __bf16* __restrict__ X, const __bf16* __restrict__ W,
                __bf16* __restrict__ Q, __bf16* __restrict__ Ko,
                __bf16* __restrict__ Vt) {
  const int lane = threadIdx.x & 31;
  const int wave = threadIdx.x >> 5;
  const int m0 = blockIdx.y * 32;
  const int n0 = blockIdx.x * 512 + wave * 64;

  v8f acc[2][4];
  #pragma unroll
  for (int i = 0; i < 2; ++i)
    #pragma unroll
    for (int j = 0; j < 4; ++j) acc[i][j] = vzero8();

  const __bf16* xp0 = X + (size_t)m0 * HID;
  const __bf16* xp1 = X + (size_t)(m0 + 16) * HID;
  for (int k = 0; k < HID; k += 32) {
    v16bf a0 = load_fragA(xp0 + k, HID);
    v16bf a1 = load_fragA(xp1 + k, HID);
    #pragma unroll
    for (int j = 0; j < 4; ++j) {
      v16bf bf = load_fragB(W + (size_t)(n0 + j * 16) * HID + k, HID);
      acc[0][j] = wmma_bf16(a0, bf, acc[0][j]);
      acc[1][j] = wmma_bf16(a1, bf, acc[1][j]);
    }
  }

  // Epilogue: D-tile layout  m = m0+16i+g+8h , n = n0+16j+(lane&15)
  const int hi8 = (lane >> 4) << 3;
  const int nl  = lane & 15;
  #pragma unroll
  for (int j = 0; j < 4; ++j) {
    const int nt = n0 + j * 16;
    const int region = nt >> 12;         // 0:Q 1:K 2:V
    const int h = (nt & 4095) >> 7;
    const int d = (nt & 127) + nl;
    #pragma unroll
    for (int i = 0; i < 2; ++i) {
      #pragma unroll
      for (int g = 0; g < 8; ++g) {
        const int m = m0 + i * 16 + g + hi8;
        const int bb = m >> 11;          // batch
        const int s  = m & 2047;         // sequence
        const float v = acc[i][j][g];
        const size_t hb = (size_t)(bb * NHEAD + h);
        if (region == 0)
          Q[(hb * SEQ + s) * HD + d] = (__bf16)(v * QSCALE);
        else if (region == 1)
          Ko[(hb * SEQ + s) * HD + d] = (__bf16)v;
        else
          Vt[(hb * HD + d) * SEQ + s] = (__bf16)v;
      }
    }
  }
}

// ---------------------------------------------------------------------------
// Causal flash attention, transposed formulation. One wave owns 16 queries.
// grid = (16 q-blocks, 64 head-batches), 256 threads (8 waves).
__global__ void __launch_bounds__(256)
attn_kernel(const __bf16* __restrict__ Q, const __bf16* __restrict__ K,
            const __bf16* __restrict__ Vt, __bf16* __restrict__ Out) {
  __shared__ __bf16 sc[8][16][128];

  const int lane = threadIdx.x & 31;
  const int wave = threadIdx.x >> 5;
  const int bh = blockIdx.y;
  const int qb = blockIdx.x * 128 + wave * 16;

  const __bf16* Qp = Q + (size_t)bh * SEQ * HD;
  const __bf16* Kp = K + (size_t)bh * SEQ * HD;
  const __bf16* Vp = Vt + (size_t)bh * HD * SEQ;

  // Q^T B-fragments (K dim = head_dim, 4 chunks of 32), kept resident.
  v16bf qf[4];
  #pragma unroll
  for (int t = 0; t < 4; ++t)
    qf[t] = load_fragB(Qp + (size_t)qb * HD + t * 32, HD);

  const int hi8 = (lane >> 4) << 3;
  const int qg = qb + (lane & 15);       // query this lane owns
  float m_run = -1e30f, l_run = 0.0f;
  v8f o[8];
  #pragma unroll
  for (int t = 0; t < 8; ++t) o[t] = vzero8();

  for (int kvb = 0; kvb < qb + 16; kvb += 32) {
    // S^T tiles: s0 covers kv {kvb+g+16h}, s1 covers kv {kvb+8+g+16h}
    v8f s0 = vzero8(), s1 = vzero8();
    const __bf16* kp0 = Kp + (size_t)kvb * HD;
    const __bf16* kp1 = Kp + (size_t)(kvb + 8) * HD;
    #pragma unroll
    for (int t = 0; t < 4; ++t) {
      v16bf kf = load_fragA_perm(kp0 + t * 32, HD);
      s0 = wmma_bf16(kf, qf[t], s0);
    }
    #pragma unroll
    for (int t = 0; t < 4; ++t) {
      v16bf kf = load_fragA_perm(kp1 + t * 32, HD);
      s1 = wmma_bf16(kf, qf[t], s1);
    }

    if (kvb + 31 > qb) {                 // diagonal step: causal mask
      const int kbase = kvb + ((lane >> 4) << 4);
      #pragma unroll
      for (int g = 0; g < 8; ++g) {
        if (kbase + g > qg)     s0[g] = NEGF;
        if (kbase + 8 + g > qg) s1[g] = NEGF;
      }
    }

    // online softmax: per-lane over 16 kv, + one half-wave swap
    float mx = m_run;
    #pragma unroll
    for (int g = 0; g < 8; ++g) mx = fmaxf(mx, fmaxf(s0[g], s1[g]));
    mx = fmaxf(mx, __shfl_xor(mx, 16, 32));
    const float alpha = __builtin_amdgcn_exp2f((m_run - mx) * L2E);

    float rs = 0.0f;
    v8f p0, p1;
    #pragma unroll
    for (int g = 0; g < 8; ++g) {
      p0[g] = __builtin_amdgcn_exp2f((s0[g] - mx) * L2E);
      p1[g] = __builtin_amdgcn_exp2f((s1[g] - mx) * L2E);
      rs += p0[g] + p1[g];
    }
    rs += __shfl_xor(rs, 16, 32);
    l_run = l_run * alpha + rs;
    m_run = mx;

    // P^T B-fragment: element e<8 <- s0[g], e>=8 <- s1[g] (by construction)
    v16bf pf;
    #pragma unroll
    for (int g = 0; g < 8; ++g) {
      pf[g]     = (__bf16)p0[g];
      pf[g + 8] = (__bf16)p1[g];
    }

    #pragma unroll
    for (int t = 0; t < 8; ++t) {
      #pragma unroll
      for (int g = 0; g < 8; ++g) o[t][g] *= alpha;
    }
    // O^T += V^T * P^T  (8 d-tiles)
    #pragma unroll
    for (int t = 0; t < 8; ++t) {
      v16bf vf = load_fragA(Vp + (size_t)(t * 16) * SEQ + kvb, SEQ);
      o[t] = wmma_bf16(vf, pf, o[t]);
    }
  }

  // Epilogue: normalize, transpose O^T[d][q] -> [q][d] via LDS, coalesced store.
  const float inv = 1.0f / l_run;
  #pragma unroll
  for (int t = 0; t < 8; ++t) {
    #pragma unroll
    for (int g = 0; g < 8; ++g)
      sc[wave][lane & 15][t * 16 + g + hi8] = (__bf16)(o[t][g] * inv);
  }
  __syncthreads();

  const size_t base = ((size_t)((bh >> 5) * SEQ + qb)) * HID + (size_t)(bh & 31) * HD;
  #pragma unroll
  for (int r = 0; r < 16; ++r) {
    u32x2 v = *(const u32x2*)&sc[wave][r][lane * 4];
    *(u32x2*)(Out + base + (size_t)r * HID + lane * 4) = v;
  }
}

// ---------------------------------------------------------------------------
// o_proj GEMM: attn[4096 x 4096] * o_w^T -> f32 output
__global__ void __launch_bounds__(256)
oproj_gemm_kernel(const __bf16* __restrict__ A, const __bf16* __restrict__ W,
                  float* __restrict__ Out) {
  const int lane = threadIdx.x & 31;
  const int wave = threadIdx.x >> 5;
  const int m0 = blockIdx.y * 32;
  const int n0 = blockIdx.x * 512 + wave * 64;

  v8f acc[2][4];
  #pragma unroll
  for (int i = 0; i < 2; ++i)
    #pragma unroll
    for (int j = 0; j < 4; ++j) acc[i][j] = vzero8();

  const __bf16* ap0 = A + (size_t)m0 * HID;
  const __bf16* ap1 = A + (size_t)(m0 + 16) * HID;
  for (int k = 0; k < HID; k += 32) {
    v16bf a0 = load_fragA(ap0 + k, HID);
    v16bf a1 = load_fragA(ap1 + k, HID);
    #pragma unroll
    for (int j = 0; j < 4; ++j) {
      v16bf bf = load_fragB(W + (size_t)(n0 + j * 16) * HID + k, HID);
      acc[0][j] = wmma_bf16(a0, bf, acc[0][j]);
      acc[1][j] = wmma_bf16(a1, bf, acc[1][j]);
    }
  }

  const int hi8 = (lane >> 4) << 3;
  const int nl  = lane & 15;
  #pragma unroll
  for (int i = 0; i < 2; ++i) {
    #pragma unroll
    for (int j = 0; j < 4; ++j) {
      #pragma unroll
      for (int g = 0; g < 8; ++g) {
        const int m = m0 + i * 16 + g + hi8;
        Out[(size_t)m * HID + n0 + j * 16 + nl] = acc[i][j][g];
      }
    }
  }
}

// ---------------------------------------------------------------------------
extern "C" void kernel_launch(void* const* d_in, const int* in_sizes, int n_in,
                              void* d_out, int out_size, void* d_ws, size_t ws_size,
                              hipStream_t stream) {
  (void)in_sizes; (void)n_in; (void)out_size; (void)ws_size;
  const float* hidden = (const float*)d_in[0];
  // d_in[1] (attention_mask) is exactly the causal mask; applied analytically.
  const float* wpack  = (const float*)d_in[2];
  const float* oprojw = (const float*)d_in[3];
  float* out = (float*)d_out;

  char* ws = (char*)d_ws;                       // 224 MiB used
  __bf16* Xb  = (__bf16*)(ws);                  //  32 MiB  X bf16
  __bf16* Wb  = (__bf16*)(ws + ((size_t)32  << 20));  // 96 MiB  W_pack bf16
  __bf16* Qb  = (__bf16*)(ws + ((size_t)128 << 20));  // 32 MiB  Q [b][h][s][d]
  __bf16* Kb  = (__bf16*)(ws + ((size_t)160 << 20));  // 32 MiB  K [b][h][s][d]
  __bf16* Vtb = (__bf16*)(ws + ((size_t)192 << 20));  // 32 MiB  V^T [b][h][d][s]
  __bf16* Ob  = Wb;   // o_proj bf16 weights: reuse W_pack slot (cast after QKV)
  __bf16* Ab  = Xb;   // attention output:    reuse X slot (written after QKV)

  const int nX4 = HID * HID / 4;                // 4,194,304
  const int nW4 = 3 * HID * HID / 4;            // 12,582,912

  cast_f32_bf16<<<dim3((nX4 + 255) / 256), 256, 0, stream>>>(hidden, Xb, nX4);
  cast_f32_bf16<<<dim3((nW4 + 255) / 256), 256, 0, stream>>>(wpack, Wb, nW4);

  qkv_gemm_kernel<<<dim3(24, 128), 256, 0, stream>>>(Xb, Wb, Qb, Kb, Vtb);

  cast_f32_bf16<<<dim3((nX4 + 255) / 256), 256, 0, stream>>>(oprojw, Ob, nX4);

  attn_kernel<<<dim3(16, 64), 256, 0, stream>>>(Qb, Kb, Vtb, Ab);

  oproj_gemm_kernel<<<dim3(8, 128), 256, 0, stream>>>(Ab, Ob, out);
}
